// LSTM_3229815407359
// MI455X (gfx1250) — matrix operand order repaired and checked by reading, not compiled, not run on hardware
//
#include <hip/hip_runtime.h>
#include <hip/hip_bf16.h>

#define T 4096
#define E 1024
#define H 1024

#define NWG 64                       // persistent workgroups for the scan
#define ROWS_PER_WG (H / NWG)        // 16 rows of h per WG
#define DOTS_PER_WG (ROWS_PER_WG*4)  // 64 dot products (4 gates)
#define DOTS_PER_WAVE (DOTS_PER_WG/8)// 8 dots per wave (8 waves/WG)

typedef __attribute__((ext_vector_type(16))) __bf16 v16bf;
typedef __attribute__((ext_vector_type(8)))  __bf16 v8bf;
typedef __attribute__((ext_vector_type(8)))  float  v8f;

// ---------------------------------------------------------------------------
// Phase 0: fp32 -> bf16 conversion of x and the 4 U matrices, plus
// deterministic re-init of h0 and the grid-barrier counter every launch.
// ---------------------------------------------------------------------------
__global__ __launch_bounds__(256)
void k_convert_init(const float* __restrict__ x,
                    const float* __restrict__ Uf, const float* __restrict__ Uc,
                    const float* __restrict__ Ug, const float* __restrict__ Uq,
                    __bf16* __restrict__ xbf, __bf16* __restrict__ ubf,
                    float* __restrict__ hbuf, int* __restrict__ counter) {
    size_t v = (size_t)blockIdx.x * blockDim.x + threadIdx.x;
    size_t i = v * 4;
    const size_t TE = (size_t)T * E;
    const size_t UE = (size_t)H * E;

    const float* src;
    __bf16* dst;
    size_t off;
    if (i < TE) {
        src = x; dst = xbf; off = i;
    } else {
        size_t j = i - TE;
        int plane = (int)(j / UE);
        off = j - (size_t)plane * UE;
        src = (plane == 0) ? Uf : (plane == 1) ? Uc : (plane == 2) ? Ug : Uq;
        dst = ubf + (size_t)plane * UE;
    }
    float4 f = *reinterpret_cast<const float4*>(src + off);
    __bf16 tmp[4] = {(__bf16)f.x, (__bf16)f.y, (__bf16)f.z, (__bf16)f.w};
    *reinterpret_cast<uint2*>(dst + off) = *reinterpret_cast<uint2*>(tmp);

    if (v < 2 * H) hbuf[v] = 0.0f;   // zero both h double-buffers
    if (v == 0)    *counter = 0;     // reset grid barrier
}

// ---------------------------------------------------------------------------
// Phase 1: pre[g][t][n] = (x @ U_g^T)[t][n] + B_g[n]   via bf16 WMMA.
// Each wave: one 16(M) x 64(N) tile, K swept 0..1024 in steps of 32.
// grid = (T/16, H/512, 4 gates), block = 256 (8 waves, wave w -> N base w*64).
// ---------------------------------------------------------------------------
__device__ inline v16bf ld_bf16_pair(const __bf16* p0, const __bf16* p1) {
    v16bf r;
    *reinterpret_cast<v8bf*>(&r)       = *reinterpret_cast<const v8bf*>(p0);
    *(reinterpret_cast<v8bf*>(&r) + 1) = *reinterpret_cast<const v8bf*>(p1);
    return r;
}

__global__ __launch_bounds__(256)
void k_gemm_pre(const __bf16* __restrict__ xbf, const __bf16* __restrict__ ubf,
                const float* __restrict__ Bf, const float* __restrict__ Bc,
                const float* __restrict__ Bg, const float* __restrict__ Bq,
                float* __restrict__ pre) {
    const int tid  = threadIdx.x;
    const int wave = tid >> 5;
    const int lane = tid & 31;
    const int idx  = lane & 15;
    const int hi   = lane >> 4;

    const int t0   = blockIdx.x * 16;
    const int nb   = blockIdx.y * 512 + wave * 64;
    const int gate = blockIdx.z;

    const __bf16* Ub   = ubf + (size_t)gate * H * E;
    const float*  bias = (gate == 0) ? Bf : (gate == 1) ? Bc : (gate == 2) ? Bg : Bq;

    // A fragment source: row t0+idx of x, 16 contiguous K at k0+16*hi
    const __bf16* arow = xbf + (size_t)(t0 + idx) * E;
    // B fragment sources: row nb+s*16+idx of U, 16 contiguous K at k0+16*hi
    const __bf16* brow0 = Ub + (size_t)(nb +  0 + idx) * E;
    const __bf16* brow1 = Ub + (size_t)(nb + 16 + idx) * E;
    const __bf16* brow2 = Ub + (size_t)(nb + 32 + idx) * E;
    const __bf16* brow3 = Ub + (size_t)(nb + 48 + idx) * E;

    v8f z = {0.f, 0.f, 0.f, 0.f, 0.f, 0.f, 0.f, 0.f};
    v8f acc0 = z, acc1 = z, acc2 = z, acc3 = z;

    for (int k0 = 0; k0 < E; k0 += 32) {
        const int ka = k0 + 16 * hi;
        const int kb = k0 + 16 * hi;
        v16bf a  = ld_bf16_pair(arow + ka, arow + ka + 8);
        v16bf b0 = ld_bf16_pair(brow0 + kb, brow0 + kb + 8);
        v16bf b1 = ld_bf16_pair(brow1 + kb, brow1 + kb + 8);
        v16bf b2 = ld_bf16_pair(brow2 + kb, brow2 + kb + 8);
        v16bf b3 = ld_bf16_pair(brow3 + kb, brow3 + kb + 8);
        acc0 = __builtin_amdgcn_wmma_f32_16x16x32_bf16(false, a, false, b0, (short)0, acc0, false, false);
        acc1 = __builtin_amdgcn_wmma_f32_16x16x32_bf16(false, a, false, b1, (short)0, acc1, false, false);
        acc2 = __builtin_amdgcn_wmma_f32_16x16x32_bf16(false, a, false, b2, (short)0, acc2, false, false);
        acc3 = __builtin_amdgcn_wmma_f32_16x16x32_bf16(false, a, false, b3, (short)0, acc3, false, false);
    }

    // Epilogue: D layout -> row = r + 8*hi, col = idx (per 16x16 subtile)
    float* pg = pre + (size_t)gate * T * H;
    v8f accs[4] = {acc0, acc1, acc2, acc3};
#pragma unroll
    for (int s = 0; s < 4; ++s) {
        const int n = nb + s * 16 + idx;
        const float b = bias[n];
#pragma unroll
        for (int r = 0; r < 8; ++r) {
            pg[(size_t)(t0 + r + 8 * hi) * H + n] = accs[s][r] + b;
        }
    }
}

// ---------------------------------------------------------------------------
// Phase 2: persistent-grid sequential scan.
// WG w owns rows [w*16, w*16+16) of h, all 4 gates. Each wave computes one
// 1024-long dot at a time (coalesced b128 W loads; h in LDS; shfl reduce).
// One agent-scope atomic barrier per timestep (monotone counter).
// ---------------------------------------------------------------------------
__device__ inline float sigm(float v) { return 1.0f / (1.0f + expf(-v)); }

__global__ __launch_bounds__(256)
void k_recurrence(const float* __restrict__ Wf, const float* __restrict__ Wc,
                  const float* __restrict__ Wg, const float* __restrict__ Wq,
                  const float* __restrict__ pre, float* __restrict__ hbuf,
                  int* __restrict__ counter, float* __restrict__ out) {
    __shared__ float4 hsh[H / 4];          // full h vector, 4 KB
    __shared__ float  preact[DOTS_PER_WG]; // W.h partial results, 256 B

    const int tid  = threadIdx.x;
    const int wg   = blockIdx.x;
    const int wave = tid >> 5;
    const int lane = tid & 31;

    float s_state = 0.0f;                  // cell state for row wg*16 + tid (tid<16)

    for (int t = 0; t < T; ++t) {
        // --- broadcast h_{t-1} into LDS ---
        hsh[tid] = *(reinterpret_cast<const float4*>(hbuf + (size_t)(t & 1) * H) + tid);
        __syncthreads();

        // cache h in registers for reuse across this wave's 8 dots
        float4 hv[8];
#pragma unroll
        for (int j = 0; j < 8; ++j) hv[j] = hsh[j * 32 + lane];

        // --- 8 dot products per wave: d -> (row, gate) ---
#pragma unroll 1
        for (int i = 0; i < DOTS_PER_WAVE; ++i) {
            const int d    = wave * DOTS_PER_WAVE + i;   // 0..63
            const int rloc = d >> 2;                     // 0..15
            const int g    = d & 3;                      // gate, wave-uniform
            const int row  = wg * ROWS_PER_WG + rloc;
            const float* wp = (g == 0) ? Wf : (g == 1) ? Wc : (g == 2) ? Wg : Wq;
            const float4* wr = reinterpret_cast<const float4*>(wp + (size_t)row * H);

            float acc = 0.0f;
#pragma unroll
            for (int j = 0; j < 8; ++j) {
                float4 wv = wr[j * 32 + lane];           // coalesced 512B/wave
                acc = fmaf(wv.x, hv[j].x, acc);
                acc = fmaf(wv.y, hv[j].y, acc);
                acc = fmaf(wv.z, hv[j].z, acc);
                acc = fmaf(wv.w, hv[j].w, acc);
            }
#pragma unroll
            for (int off = 16; off >= 1; off >>= 1)
                acc += __shfl_xor(acc, off, 32);
            if (lane == 0) preact[d] = acc;
        }
        __syncthreads();

        // --- gate nonlinearities + state update for this WG's 16 rows ---
        if (tid < ROWS_PER_WG) {
            const int row = wg * ROWS_PER_WG + tid;
            const size_t tH = (size_t)t * H;
            const float af = preact[tid * 4 + 0] + pre[0 * (size_t)T * H + tH + row];
            const float ac = preact[tid * 4 + 1] + pre[1 * (size_t)T * H + tH + row];
            const float ag = preact[tid * 4 + 2] + pre[2 * (size_t)T * H + tH + row];
            const float aq = preact[tid * 4 + 3] + pre[3 * (size_t)T * H + tH + row];
            const float f = sigm(af), c = sigm(ac), gg = sigm(ag), q = sigm(aq);
            s_state = f * s_state + gg * c;
            const float h = tanhf(s_state) * q;
            hbuf[(size_t)((t + 1) & 1) * H + row] = h;
            out[tH + row] = h;
        }
        __threadfence();     // make h slice visible at device scope
        __syncthreads();

        // --- grid barrier: monotone counter, one arrival per WG ---
        if (tid == 0) {
            __hip_atomic_fetch_add(counter, 1, __ATOMIC_ACQ_REL, __HIP_MEMORY_SCOPE_AGENT);
            const int target = NWG * (t + 1);
            while (__hip_atomic_load(counter, __ATOMIC_ACQUIRE, __HIP_MEMORY_SCOPE_AGENT) < target)
                __builtin_amdgcn_s_sleep(1);
        }
        __syncthreads();
        __threadfence();
    }
}

// ---------------------------------------------------------------------------
extern "C" void kernel_launch(void* const* d_in, const int* in_sizes, int n_in,
                              void* d_out, int out_size, void* d_ws, size_t ws_size,
                              hipStream_t stream) {
    const float* x  = (const float*)d_in[0];
    const float* Uf = (const float*)d_in[1];
    const float* Wf = (const float*)d_in[2];
    const float* Bf = (const float*)d_in[3];
    const float* Ug = (const float*)d_in[4];
    const float* Wg = (const float*)d_in[5];
    const float* Bg = (const float*)d_in[6];
    const float* Uq = (const float*)d_in[7];
    const float* Wq = (const float*)d_in[8];
    const float* Bq = (const float*)d_in[9];
    const float* Uc = (const float*)d_in[10];
    const float* Wc = (const float*)d_in[11];
    const float* Bc = (const float*)d_in[12];
    float* out = (float*)d_out;

    // workspace layout: [pre 64MB][xbf 8MB][ubf 8MB][hbuf 8KB][counter]
    float*  pre     = (float*)d_ws;
    __bf16* xbf     = (__bf16*)((char*)d_ws + (size_t)4 * T * H * sizeof(float));
    __bf16* ubf     = xbf + (size_t)T * E;
    float*  hbuf    = (float*)(ubf + (size_t)4 * H * E);
    int*    counter = (int*)(hbuf + 2 * H);

    // phase 0: bf16 conversion + state init (8,388,608 elems / 4 per thread)
    const int conv_blocks = (T * E + 4 * H * E) / 4 / 256;
    k_convert_init<<<conv_blocks, 256, 0, stream>>>(x, Uf, Uc, Ug, Uq,
                                                    xbf, ubf, hbuf, counter);
    // phase 1: 4 input-projection GEMMs via bf16 WMMA
    k_gemm_pre<<<dim3(T / 16, 2, 4), 256, 0, stream>>>(xbf, ubf, Bf, Bc, Bg, Bq, pre);
    // phase 2: persistent sequential scan
    k_recurrence<<<NWG, 256, 0, stream>>>(Wf, Wc, Wg, Wq, pre, hbuf, counter, out);
}
